// InteractionGNNBlock_23974507446585
// MI455X (gfx1250) — matrix operand order repaired and checked by reading, not compiled, not run on hardware
//
#include <hip/hip_runtime.h>

// ---------------- problem constants (from reference) ----------------
#define NN  100000   // nodes
#define NE  500000   // edges
#define NF  12       // node feature dim
#define DM  64       // d_model == d_hidden

typedef __attribute__((ext_vector_type(16))) _Float16 v16h;
typedef __attribute__((ext_vector_type(8)))  float    v8f;

union Frag16 { v16h v; _Float16 h[16]; unsigned int u[8]; };

// Exact-GELU via branch-free erf (Abramowitz-Stegun 7.1.26, |err| <= 1.5e-7,
// far below the f16-WMMA quantization error of the matmuls).
__device__ __forceinline__ float gelu_exact(float x) {
    const float z = __builtin_fabsf(x) * 0.70710678118654752440f;
    const float t = __builtin_amdgcn_rcpf(1.0f + 0.3275911f * z);
    float poly =         1.061405429f;
    poly = poly * t -    1.453152027f;
    poly = poly * t +    1.421413741f;
    poly = poly * t -    0.284496736f;
    poly = poly * t +    0.254829592f;
    poly = poly * t;
    const float e       = __expf(-z * z);
    const float erf_abs = 1.0f - poly * e;
    const float erf_v   = __builtin_copysignf(erf_abs, x);
    return 0.5f * x * (1.0f + erf_v);
}

// pack two f32 -> one dword of two f16
__device__ __forceinline__ unsigned int pkh(float a, float b) {
    union { _Float16 h[2]; unsigned int u; } t;
    t.h[0] = (_Float16)a; t.h[1] = (_Float16)b;
    return t.u;
}

// A-layout column permutation within a 32-col chunk: {0-7,16-23,8-15,24-31}
__device__ __forceinline__ int swz32(int c) {
    return (c & 7) | ((c & 16) >> 1) | ((c & 8) << 1);
}

// Fused 2-layer MLP: out = (GELU(X @ W1 + b1)) @ W2 + b2
//   MODE 0: X[r] = x0[r, 0:Kact]                            (node encoder)
//   MODE 1: X[r] = concat(x0[sidx[r]], x0[didx[r]])         (edge encoder)
//   MODE 2: X[r] = concat(x0[r], x1[r])                     (node network)
//   MODE 3: X[r] = concat(x0[sidx[r]], x0[didx[r]], x2[r])  (edge network)
template<int KP, int MODE>
__global__ __launch_bounds__(256)
void mlp2_kernel(const float* __restrict__ x0, const float* __restrict__ x1,
                 const float* __restrict__ x2,
                 const int* __restrict__ sidx, const int* __restrict__ didx,
                 const float* __restrict__ W1, const float* __restrict__ B1,
                 const float* __restrict__ W2, const float* __restrict__ B2,
                 float* __restrict__ out, int rows, int Kact)
{
    // weights pre-swizzled into B-fragment layout: [kchunk][ntile][lane][v2] dwords
    __shared__ __align__(16) unsigned int sW1s[KP * 32];      // KP*64 f16
    __shared__ __align__(16) unsigned int sW2s[64 * 32];      // 64*64 f16
    // input chunk, rows in A-fragment layout: row-major, 16 dwords/row (swizzled)
    __shared__ __align__(16) unsigned int sXs[128 * 16];
    // hidden activations, swizzled per 32-col chunk, f16
    __shared__ __align__(16) _Float16 sHs[128 * DM];
    __shared__ float sB1[DM], sB2[DM];

    const int tid  = threadIdx.x;
    const int lane = tid & 31;
    const int wave = tid >> 5;
    const int lrow = lane & 15;
    const int half = lane >> 4;

    // ---- stage W1 into fragment layout (coalesced global reads) ----
    for (int i = tid; i < KP * 32; i += 256) {
        const int p = i >> 6;            // k-pair index
        const int n = i & 63;            // output column
        const int k = p << 1;
        const float lo = (k     < Kact) ? W1[k * DM + n]       : 0.0f;
        const float hi = (k + 1 < Kact) ? W1[(k + 1) * DM + n] : 0.0f;
        const int ck = k >> 5, kk = k & 31;
        const int hf = (kk >> 3) & 1, v2 = ((kk & 7) >> 1) + ((kk >> 4) << 2);
        const int nt = n >> 4, lr = n & 15;
        sW1s[(((ck << 2) + nt) << 8) + ((hf << 4) + lr) * 8 + v2] = pkh(lo, hi);
    }
    // ---- stage W2 (64x64) ----
    for (int i = tid; i < 64 * 32; i += 256) {
        const int p = i >> 6;
        const int n = i & 63;
        const int k = p << 1;
        const float lo = W2[k * DM + n];
        const float hi = W2[(k + 1) * DM + n];
        const int ck = k >> 5, kk = k & 31;
        const int hf = (kk >> 3) & 1, v2 = ((kk & 7) >> 1) + ((kk >> 4) << 2);
        const int nt = n >> 4, lr = n & 15;
        sW2s[(((ck << 2) + nt) << 8) + ((hf << 4) + lr) * 8 + v2] = pkh(lo, hi);
    }
    if (tid < DM) { sB1[tid] = B1[tid]; sB2[tid] = B2[tid]; }
    __syncthreads();

    const v8f vzero = {0.f, 0.f, 0.f, 0.f, 0.f, 0.f, 0.f, 0.f};
    const int ntiles = (rows + 127) >> 7;

    for (int tile = blockIdx.x; tile < ntiles; tile += gridDim.x) {
        const int rbase = tile << 7;
        v8f acc[4];
        acc[0] = vzero; acc[1] = vzero; acc[2] = vzero; acc[3] = vzero;

        // ---------------- layer 1: K-chunk loop ----------------
        for (int k0 = 0; k0 < KP; k0 += 32) {
            __syncthreads();   // previous sX contents consumed
            {   // assemble sX: thread handles one half-row (16 consecutive cols)
                const int rl = tid >> 1;
                const int cb = (tid & 1) << 4;
                const int r  = rbase + rl;
                float vals[16];
                if (r < rows) {
                    int s = 0, d = 0;
                    if (MODE == 1 || MODE == 3) { s = sidx[r]; d = didx[r]; }
                    #pragma unroll
                    for (int j = 0; j < 16; ++j) {
                        const int c = k0 + cb + j;
                        float v;
                        if (MODE == 0) {
                            v = (c < Kact) ? x0[(long)r * Kact + c] : 0.0f;
                        } else if (MODE == 1) {
                            v = (c < DM) ? x0[(long)s * DM + c]
                                         : x0[(long)d * DM + (c - DM)];
                        } else if (MODE == 2) {
                            v = (c < DM) ? x0[(long)r * DM + c]
                                         : x1[(long)r * DM + (c - DM)];
                        } else {
                            v = (c < DM)     ? x0[(long)s * DM + c]
                              : (c < 2 * DM) ? x0[(long)d * DM + (c - DM)]
                                             : x2[(long)r * DM + (c - 2 * DM)];
                        }
                        vals[j] = v;
                    }
                } else {
                    #pragma unroll
                    for (int j = 0; j < 16; ++j) vals[j] = 0.0f;
                }
                // cb=0 -> dwords {0..3, 8..11}; cb=16 -> dwords {4..7, 12..15}
                uint4 qa, qb;
                qa.x = pkh(vals[0],  vals[1]);  qa.y = pkh(vals[2],  vals[3]);
                qa.z = pkh(vals[4],  vals[5]);  qa.w = pkh(vals[6],  vals[7]);
                qb.x = pkh(vals[8],  vals[9]);  qb.y = pkh(vals[10], vals[11]);
                qb.z = pkh(vals[12], vals[13]); qb.w = pkh(vals[14], vals[15]);
                const int base0 = rl * 16 + (cb >> 2);
                *(uint4*)&sXs[base0]     = qa;
                *(uint4*)&sXs[base0 + 8] = qb;
            }
            __syncthreads();

            // A fragment: 2 x 16B LDS loads (pre-swizzled rows)
            Frag16 a;
            const int arow = (wave << 4) + lrow;
            {
                const unsigned int* ap = &sXs[arow * 16 + half * 8];
                *(uint4*)&a.u[0] = *(const uint4*)ap;
                *(uint4*)&a.u[4] = *(const uint4*)(ap + 4);
            }
            const int ck = k0 >> 5;
            #pragma unroll
            for (int nt = 0; nt < 4; ++nt) {
                Frag16 b;
                const unsigned int* bp = &sW1s[(((ck << 2) + nt) << 8) + lane * 8];
                *(uint4*)&b.u[0] = *(const uint4*)bp;
                *(uint4*)&b.u[4] = *(const uint4*)(bp + 4);
                acc[nt] = __builtin_amdgcn_wmma_f32_16x16x32_f16(
                    false, a.v, false, b.v, (short)0, acc[nt], false, false);
            }
        }

        // ---------------- bias + GELU -> sHs (swizzled, per-wave slice) ----------------
        #pragma unroll
        for (int nt = 0; nt < 4; ++nt) {
            const int n    = (nt << 4) + lrow;
            const int hpos = (n & 32) + swz32(n & 31);
            #pragma unroll
            for (int rr = 0; rr < 8; ++rr) {
                const float xv = acc[nt][rr] + sB1[n];
                sHs[((wave << 4) + rr + (half << 3)) * DM + hpos] =
                    (_Float16)gelu_exact(xv);
            }
        }

        // ---------------- layer 2 ----------------
        v8f acc2[4];
        acc2[0] = vzero; acc2[1] = vzero; acc2[2] = vzero; acc2[3] = vzero;
        const int arow2 = (wave << 4) + lrow;
        const unsigned int* sHd = (const unsigned int*)sHs;
        #pragma unroll
        for (int k0 = 0; k0 < DM; k0 += 32) {
            Frag16 a;
            const unsigned int* ap = &sHd[arow2 * 32 + (k0 >> 1) + half * 8];
            *(uint4*)&a.u[0] = *(const uint4*)ap;
            *(uint4*)&a.u[4] = *(const uint4*)(ap + 4);
            const int ck = k0 >> 5;
            #pragma unroll
            for (int nt = 0; nt < 4; ++nt) {
                Frag16 b;
                const unsigned int* bp = &sW2s[(((ck << 2) + nt) << 8) + lane * 8];
                *(uint4*)&b.u[0] = *(const uint4*)bp;
                *(uint4*)&b.u[4] = *(const uint4*)(bp + 4);
                acc2[nt] = __builtin_amdgcn_wmma_f32_16x16x32_f16(
                    false, a.v, false, b.v, (short)0, acc2[nt], false, false);
            }
        }

        // ---------------- bias + store f32 ----------------
        #pragma unroll
        for (int nt = 0; nt < 4; ++nt) {
            const int n = (nt << 4) + lrow;
            #pragma unroll
            for (int rr = 0; rr < 8; ++rr) {
                const int row = rbase + (wave << 4) + rr + (half << 3);
                if (row < rows)
                    out[(long)row * DM + n] = acc2[nt][rr] + sB2[n];
            }
        }
    }
}

__global__ __launch_bounds__(256)
void zero_kernel(float4* __restrict__ p, long n4) {
    const float4 z = {0.f, 0.f, 0.f, 0.f};
    for (long i = (long)blockIdx.x * 256 + threadIdx.x; i < n4; i += (long)gridDim.x * 256)
        p[i] = z;
}

__global__ __launch_bounds__(256)
void scatter_add_kernel(const float* __restrict__ edges, const int* __restrict__ dst,
                        float* __restrict__ msg, int E) {
    const long total = (long)E * 32;   // one thread per 2 columns
    for (long i = (long)blockIdx.x * 256 + threadIdx.x; i < total; i += (long)gridDim.x * 256) {
        const int e = (int)(i >> 5);
        const int c = (int)(i & 31) << 1;
        const float2 v = *(const float2*)&edges[((long)e << 6) + c];
        float* m = &msg[((long)dst[e] << 6) + c];
        unsafeAtomicAdd(m,     v.x);
        unsafeAtomicAdd(m + 1, v.y);
    }
}

static inline int imin(int a, int b) { return a < b ? a : b; }

extern "C" void kernel_launch(void* const* d_in, const int* in_sizes, int n_in,
                              void* d_out, int out_size, void* d_ws, size_t ws_size,
                              hipStream_t stream) {
    (void)in_sizes; (void)n_in; (void)out_size; (void)ws_size;

    const float* node_attr = (const float*)d_in[0];
    const int*   graph     = (const int*)d_in[1];
    const int*   src = graph;
    const int*   dst = graph + NE;

    // params flattened in setup_inputs() insertion order (DFS):
    // node_encoder{W1,b1,W2,b2}, edge_encoder{...}, cells[i]{node_network{...}, edge_network{...}}
    const float* p[32];
    for (int i = 0; i < 32; ++i) p[i] = (const float*)d_in[2 + i];

    float* nodes = (float*)d_out;                    // [NN, 64]
    float* edges = nodes + (size_t)NN * DM;          // [NE, 64]
    float* msg   = (float*)d_ws;                     // [NN, 64] scratch

    const int nblocks = imin((NN + 127) / 128, 1024);
    const int eblocks = imin((NE + 127) / 128, 2048);

    // 1) node encoder: [NN,12] -> [NN,64]
    mlp2_kernel<32, 0><<<nblocks, 256, 0, stream>>>(
        node_attr, nullptr, nullptr, nullptr, nullptr,
        p[0], p[1], p[2], p[3], nodes, NN, NF);

    // 2) edge encoder: concat(nodes[src], nodes[dst]) -> [NE,64]
    mlp2_kernel<128, 1><<<eblocks, 256, 0, stream>>>(
        nodes, nullptr, nullptr, src, dst,
        p[4], p[5], p[6], p[7], edges, NE, 128);

    // 3) message-passing iterations
    for (int it = 0; it < 3; ++it) {
        const float* const* c = &p[8 + it * 8];

        zero_kernel<<<2048, 256, 0, stream>>>((float4*)msg, (long)NN * 16);
        scatter_add_kernel<<<4096, 256, 0, stream>>>(edges, dst, msg, NE);

        // nodes = MLP([nodes, msg])   (in-place safe: row i reads only row i)
        mlp2_kernel<128, 2><<<nblocks, 256, 0, stream>>>(
            nodes, msg, nullptr, nullptr, nullptr,
            c[0], c[1], c[2], c[3], nodes, NN, 128);

        // edges = MLP([nodes[src], nodes[dst], edges])   (in-place safe per edge row)
        mlp2_kernel<192, 3><<<eblocks, 256, 0, stream>>>(
            nodes, nullptr, edges, src, dst,
            c[4], c[5], c[6], c[7], edges, NE, 192);
    }
}
